// Tokenizer_51762945851627
// MI455X (gfx1250) — matrix operand order repaired and checked by reading, not compiled, not run on hardware
//
#include <hip/hip_runtime.h>
#include <hip/hip_bf16.h>

typedef __attribute__((ext_vector_type(16))) _Float16 v16h;
typedef __attribute__((ext_vector_type(8)))  float    v8f;
typedef __attribute__((ext_vector_type(4)))  int      i4v;

union FragU { v16h v; unsigned int u[8]; };

#define NBATCH 128      // bs*t
#define PIX    256      // h*w per batch
#define EMB    256
#define ZCH    256
#define VOCAB  4096
#define NPIX   32768    // NBATCH*PIX

// ---------------------------------------------------------------------------
// CDNA5 async global->LDS copy (ASYNCcnt path), with portable fallback.
// Builtin signature (from hipcc diagnostic): first arg is
// 'int __vector(4) __device__ *'  == addrspace(1) int4*.
// ---------------------------------------------------------------------------
#if __has_builtin(__builtin_amdgcn_global_load_async_to_lds_b128)
#define HAVE_ASYNC 1
__device__ __forceinline__ void async_copy16(const void* g, void* l) {
  __builtin_amdgcn_global_load_async_to_lds_b128(
      (__attribute__((address_space(1))) i4v*)g,
      (__attribute__((address_space(3))) i4v*)l, 0, 0);
}
#if __has_builtin(__builtin_amdgcn_s_wait_asynccnt)
#define WAIT_ASYNC(n) __builtin_amdgcn_s_wait_asynccnt(n)
#else
#define WAIT_ASYNC(n) asm volatile("s_wait_asynccnt %0" ::"n"(n) : "memory")
#endif
#else
#define HAVE_ASYNC 0
__device__ __forceinline__ void async_copy16(const void* g, void* l) {
  *(int4*)l = *(const int4*)g;
}
#define WAIT_ASYNC(n) ((void)0)
#endif

// A-matrix (16x32 f16) K-pair start for vgpr j, lane-half hi
__device__ __forceinline__ int kpA(int j, int hi) {
  return (j < 4) ? (8 * hi + 2 * j) : (16 + 8 * hi + 2 * (j - 4));
}
// B-matrix (32x16 f16) K-pair start for vgpr j, lane-half hi
__device__ __forceinline__ int kpB(int j, int hi) { return kpA(j, hi); }

__device__ __forceinline__ v8f wmma_f16(v16h a, v16h b, v8f c) {
  return __builtin_amdgcn_wmma_f32_16x16x32_f16(false, a, false, b, (short)0, c,
                                                false, false);
}

// ---------------------------------------------------------------------------
// Kernel 1: z_b = pre_w @ (2*x_b - 1) + pre_b   (per-batch 256x256x256 GEMM)
// (B needs a value transform, so staging stays on the VALU path here.)
// grid (128 batches, 4 m-blocks of 64 rows), block 256 (8 waves)
// ---------------------------------------------------------------------------
__global__ void __launch_bounds__(256)
pre_gemm_kernel(const float* __restrict__ x, const float* __restrict__ pre_w,
                const float* __restrict__ pre_b, float* __restrict__ z_out,
                _Float16* __restrict__ zf16) {
  const int b  = blockIdx.x;
  const int m0 = blockIdx.y * 64;
  __shared__ _Float16 Atile[64 * 32];    // [m][k]
  __shared__ _Float16 Btile[256 * 32];   // [p][k] (k-contiguous for B frags)
  const int tid = threadIdx.x;
  const int lane = tid & 31, w = tid >> 5;
  const int mt = w & 3, ng = w >> 2;
  const int ml = lane & 15, hi = lane >> 4;

  v8f acc[8];
#pragma unroll
  for (int t = 0; t < 8; ++t)
#pragma unroll
    for (int q = 0; q < 8; ++q) acc[t][q] = 0.0f;

  const float* xb = x + (size_t)b * (ZCH * PIX);

  for (int kk = 0; kk < 8; ++kk) {
    const int k0 = kk * 32;
    for (int i = tid; i < 64 * 32; i += 256) {
      int mm = i >> 5, kq = i & 31;
      Atile[i] = (_Float16)pre_w[(size_t)(m0 + mm) * ZCH + k0 + kq];
    }
    for (int i = tid; i < 32 * 256; i += 256) {
      int kq = i >> 8, pp = i & 255;
      Btile[pp * 32 + kq] =
          (_Float16)(2.0f * xb[(size_t)(k0 + kq) * PIX + pp] - 1.0f);
    }
    __syncthreads();

    FragU a;
#pragma unroll
    for (int j = 0; j < 8; ++j)
      a.u[j] = *(const unsigned int*)&Atile[(mt * 16 + ml) * 32 + kpA(j, hi)];
#pragma unroll
    for (int t = 0; t < 8; ++t) {
      const int nt = ng * 8 + t;
      FragU bb;
#pragma unroll
      for (int j = 0; j < 8; ++j)
        bb.u[j] = *(const unsigned int*)&Btile[(nt * 16 + ml) * 32 + kpB(j, hi)];
      acc[t] = wmma_f16(a.v, bb.v, acc[t]);
    }
    __syncthreads();
  }

#pragma unroll
  for (int t = 0; t < 8; ++t) {
    const int nt = ng * 8 + t;
    const int p = nt * 16 + ml;
#pragma unroll
    for (int i = 0; i < 8; ++i) {
      const int e = m0 + mt * 16 + i + 8 * hi;
      float v = acc[t][i] + pre_b[e];
      z_out[(size_t)b * (EMB * PIX) + (size_t)e * PIX + p] = v;
      zf16[(size_t)b * (PIX * EMB) + (size_t)p * EMB + e] = (_Float16)v;
    }
  }
}

// ---------------------------------------------------------------------------
// Kernel 2: codebook f32 -> f16 copy + per-code squared norm
// ---------------------------------------------------------------------------
__global__ void __launch_bounds__(256)
cb_prep_kernel(const float* __restrict__ cb, _Float16* __restrict__ cb16,
               float* __restrict__ csq) {
  __shared__ float red[256];
  const int v = blockIdx.x, tid = threadIdx.x;
  float val = cb[(size_t)v * EMB + tid];
  cb16[(size_t)v * EMB + tid] = (_Float16)val;
  red[tid] = val * val;
  __syncthreads();
  for (int off = 128; off > 0; off >>= 1) {
    if (tid < off) red[tid] += red[tid + off];
    __syncthreads();
  }
  if (tid == 0) csq[v] = red[0];
}

// ---------------------------------------------------------------------------
// Kernel 2b: post_w f32 -> f16
// ---------------------------------------------------------------------------
__global__ void __launch_bounds__(256)
wconv_kernel(const float* __restrict__ post_w, _Float16* __restrict__ pw16) {
  const int i = blockIdx.x * 256 + threadIdx.x;
  pw16[i] = (_Float16)post_w[i];
}

// ---------------------------------------------------------------------------
// Kernel 3: dist GEMM (32768 x 4096 x 256) + argmin -> tokens
// Codebook tiles (8 KB) staged into LDS with async global->LDS copies,
// double-buffered; each wave owns a 16-pixel M-tile with A frags in VGPRs.
// grid 256, block 256 (8 waves)
// ---------------------------------------------------------------------------
__global__ void __launch_bounds__(256)
dist_argmin_kernel(const _Float16* __restrict__ zf16,
                   const _Float16* __restrict__ cb16,
                   const float* __restrict__ csq, int* __restrict__ tok) {
  __shared__ _Float16 cbt[2][16 * 256];   // 2 x 8KB code tiles
  const int tid = threadIdx.x;
  const int lane = tid & 31, w = tid >> 5;
  const int ml = lane & 15, hi = lane >> 4;
  const int nbase = (blockIdx.x * 8 + w) * 16;

  FragU af[8];
#pragma unroll
  for (int kk = 0; kk < 8; ++kk)
#pragma unroll
    for (int j = 0; j < 8; ++j)
      af[kk].u[j] = *(const unsigned int*)&zf16[(size_t)(nbase + ml) * EMB +
                                                kk * 32 + kpA(j, hi)];

  float bestV[8];
  int bestI[8];
#pragma unroll
  for (int i = 0; i < 8; ++i) { bestV[i] = 3.4e38f; bestI[i] = 0; }

  auto stage = [&](int vt, int buf) {   // copy 8KB: 2 x 16B per thread
    const char* g = (const char*)(cb16 + (size_t)vt * 16 * EMB);
    char* l = (char*)&cbt[buf][0];
    async_copy16(g + tid * 16, l + tid * 16);
    async_copy16(g + 4096 + tid * 16, l + 4096 + tid * 16);
  };

  stage(0, 0);
  stage(1, 1);
  for (int vt = 0; vt < VOCAB / 16; ++vt) {
    if (vt < VOCAB / 16 - 1) { WAIT_ASYNC(2); } else { WAIT_ASYNC(0); }
    __syncthreads();
    const _Float16* cbl = &cbt[vt & 1][0];

    v8f acc;
#pragma unroll
    for (int q = 0; q < 8; ++q) acc[q] = 0.0f;
#pragma unroll
    for (int kk = 0; kk < 8; ++kk) {
      FragU bb;
#pragma unroll
      for (int j = 0; j < 8; ++j)
        bb.u[j] =
            *(const unsigned int*)&cbl[ml * EMB + kk * 32 + kpB(j, hi)];
      acc = wmma_f16(af[kk].v, bb.v, acc);
    }
    const float cs = csq[vt * 16 + ml];
    const int vidx = vt * 16 + ml;
#pragma unroll
    for (int i = 0; i < 8; ++i) {
      float d = cs - 2.0f * acc[i];
      if (d < bestV[i]) { bestV[i] = d; bestI[i] = vidx; }
    }
    __syncthreads();
    if (vt + 2 < VOCAB / 16) stage(vt + 2, vt & 1);
  }

#pragma unroll
  for (int i = 0; i < 8; ++i) {
    float bv = bestV[i];
    int bi = bestI[i];
#pragma unroll
    for (int off = 1; off < 16; off <<= 1) {
      float ov = __shfl_xor(bv, off, 32);
      int oi = __shfl_xor(bi, off, 32);
      if (ov < bv || (ov == bv && oi < bi)) { bv = ov; bi = oi; }
    }
    bestV[i] = bv; bestI[i] = bi;
  }
  if (ml == 0) {
#pragma unroll
    for (int i = 0; i < 8; ++i) tok[nbase + hi * 8 + i] = bestI[i];
  }
}

// ---------------------------------------------------------------------------
// Kernel 4: zq gather. grid (p=256, b=128), threads = e (coalesced codebook
// read + coalesced zq16 [b][p][e] write).
// ---------------------------------------------------------------------------
__global__ void __launch_bounds__(256)
gather_kernel(const float* __restrict__ cb, const int* __restrict__ tok,
              float* __restrict__ zq_out, _Float16* __restrict__ zq16,
              float* __restrict__ tok_out) {
  const int p = blockIdx.x, b = blockIdx.y, e = threadIdx.x;
  const int t = tok[b * PIX + p];
  const float v = cb[(size_t)t * EMB + e];
  zq_out[(size_t)b * (EMB * PIX) + (size_t)e * PIX + p] = v;
  // scale into healthy f16 range (codebook entries are ~±2.4e-4)
  zq16[(size_t)b * (PIX * EMB) + (size_t)p * EMB + e] = (_Float16)(v * 4096.0f);
  if (e == 0) tok_out[b * PIX + p] = (float)t;
}

// ---------------------------------------------------------------------------
// Kernel 5: recon_b = post_w @ zq_b + post_b.
// zq16 is [b][p][e] so B fragments load directly from global (pairs contig);
// A (post_w f16) chunks staged via async global->LDS, double-buffered.
// grid (128, 4), block 256 (8 waves)
// ---------------------------------------------------------------------------
__global__ void __launch_bounds__(256)
post_gemm_kernel(const _Float16* __restrict__ zq16,
                 const _Float16* __restrict__ pw16,
                 const float* __restrict__ post_b,
                 float* __restrict__ recon_out) {
  __shared__ _Float16 At[2][64 * 32];   // 2 x 4KB A chunks
  const int b  = blockIdx.x;
  const int m0 = blockIdx.y * 64;
  const int tid = threadIdx.x;
  const int lane = tid & 31, w = tid >> 5;
  const int mt = w & 3, ng = w >> 2;
  const int ml = lane & 15, hi = lane >> 4;

  v8f acc[8];
#pragma unroll
  for (int t = 0; t < 8; ++t)
#pragma unroll
    for (int q = 0; q < 8; ++q) acc[t][q] = 0.0f;

  const _Float16* zb = zq16 + (size_t)b * (PIX * EMB);

  auto stageA = [&](int kk, int buf) {   // 4KB: 16B per thread
    const char* g = (const char*)pw16 + (size_t)(m0 + (tid >> 2)) * (EMB * 2) +
                    kk * 64 + (tid & 3) * 16;
    char* l = (char*)&At[buf][0] + tid * 16;
    async_copy16(g, l);
  };

  stageA(0, 0);
  stageA(1, 1);
  for (int kk = 0; kk < 8; ++kk) {
    if (kk < 7) { WAIT_ASYNC(1); } else { WAIT_ASYNC(0); }
    __syncthreads();

    FragU a;
#pragma unroll
    for (int j = 0; j < 8; ++j)
      a.u[j] =
          *(const unsigned int*)&At[kk & 1][(mt * 16 + ml) * 32 + kpA(j, hi)];
#pragma unroll
    for (int t = 0; t < 8; ++t) {
      const int nt = ng * 8 + t;
      const int p = nt * 16 + ml;
      FragU bb;
#pragma unroll
      for (int j = 0; j < 8; ++j)
        bb.u[j] = *(const unsigned int*)&zb[(size_t)p * EMB + kk * 32 +
                                            kpB(j, hi)];
      acc[t] = wmma_f16(a.v, bb.v, acc[t]);
    }
    __syncthreads();
    if (kk + 2 < 8) stageA(kk + 2, kk & 1);
  }

#pragma unroll
  for (int t = 0; t < 8; ++t) {
    const int nt = ng * 8 + t;
    const int p = nt * 16 + ml;
#pragma unroll
    for (int i = 0; i < 8; ++i) {
      const int c = m0 + mt * 16 + i + 8 * hi;
      recon_out[(size_t)b * (ZCH * PIX) + (size_t)c * PIX + p] =
          acc[t][i] * (1.0f / 4096.0f) + post_b[c];
    }
  }
}

// ---------------------------------------------------------------------------
extern "C" void kernel_launch(void* const* d_in, const int* in_sizes, int n_in,
                              void* d_out, int out_size, void* d_ws,
                              size_t ws_size, hipStream_t stream) {
  const float* x      = (const float*)d_in[0];
  const float* cb     = (const float*)d_in[1];
  const float* pre_w  = (const float*)d_in[2];
  const float* pre_b  = (const float*)d_in[3];
  const float* post_w = (const float*)d_in[4];
  const float* post_b = (const float*)d_in[5];

  float* out       = (float*)d_out;
  float* z_out     = out;                     // 128*256*256
  float* zq_out    = out + 8388608;
  float* recon_out = out + 16777216;
  float* tok_out   = out + 25165824;          // 32768 tokens as float

  char* ws = (char*)d_ws;
  _Float16* zf16 = (_Float16*)(ws);                          // 16 MB [n][e]
  _Float16* cb16 = (_Float16*)(ws + (16u << 20));            // 2 MB
  float*    csq  = (float*)(ws + (16u << 20) + (2u << 20));  // 16 KB
  int*      tok  = (int*)(ws + (16u << 20) + (2u << 20) + 16384);
  _Float16* zq16 = (_Float16*)(ws + (16u << 20) + (2u << 20) + 16384 + 131072);
  _Float16* pw16 = (_Float16*)(ws + (16u << 20) + (2u << 20) + 16384 + 131072 +
                               (16u << 20));   // 128 KB

  pre_gemm_kernel<<<dim3(NBATCH, 4), 256, 0, stream>>>(x, pre_w, pre_b, z_out,
                                                       zf16);
  cb_prep_kernel<<<VOCAB, 256, 0, stream>>>(cb, cb16, csq);
  wconv_kernel<<<ZCH * EMB / 256, 256, 0, stream>>>(post_w, pw16);
  dist_argmin_kernel<<<NPIX / (16 * 8), 256, 0, stream>>>(zf16, cb16, csq, tok);
  gather_kernel<<<dim3(PIX, NBATCH), 256, 0, stream>>>(cb, tok, zq_out, zq16,
                                                       tok_out);
  post_gemm_kernel<<<dim3(NBATCH, 4), 256, 0, stream>>>(zq16, pw16, post_b,
                                                        recon_out);
}
